// MaskedLSTMBlock_47399259078708
// MI455X (gfx1250) — compile-verified, matrix operand order checked by reading, not run on hardware
//
#include <hip/hip_runtime.h>

// ---------------------------------------------------------------------------
// Masked LSTM for MI455X (gfx1250), wave32 + WMMA bf16, latency-optimized.
//
// Persistent kernel: 32 workgroups x 256 threads (8 waves), one per WGP.
// WG k owns hidden columns [16k, 16k+16)  => 64 gate columns (i,f,g,o x 16).
//
// LDS (dynamic, ~286 KB):
//   - bswu : bf16 [64][1032]  concat(w_ih,w_hh) slice, resident all steps
//   - asu  : bf16 [64][1032]  A panel [x_t | m*h_{t-1}]  (bf16)
//   - gs   : f32  [64][68]    gate accumulators
//   - cs   : f32  [64][16]    persistent cell state slice
//
// Step critical path: h bcast store -> fence/atomic arrive -> h-half GEMM.
// The x-half GEMM (no h dependence) executes inside the barrier-wait window.
// ---------------------------------------------------------------------------

typedef __attribute__((ext_vector_type(16))) __bf16 bf16x16;
typedef __attribute__((ext_vector_type(8)))  float  f32x8;

#define NWG    32
#define NTHR   256
#define TSTEPS 1024
#define BATCH  64
#define HID    512

// Row strides in elements; byte stride 2064 = 16 mod 256 -> the 16 lanes'
// 16B ds_load_b128 fragment reads cover all 64 LDS banks conflict-free.
#define WSTRIDE 1032
#define ASTRIDE 1032
#define GSTRIDE 68

#define BSW_BYTES (64 * WSTRIDE * 2)       // 132096
#define AS_BYTES  (64 * ASTRIDE * 2)       // 132096
#define GS_BYTES  (64 * GSTRIDE * 4)       // 17408
#define CS_BYTES  (64 * 16 * 4)            // 4096
#define MISC_BYTES (64*4 /*bsum*/ + 64*4 /*dm*/ + 64*4 /*mn*/)
#define SMEM_BYTES (BSW_BYTES + AS_BYTES + GS_BYTES + CS_BYTES + MISC_BYTES)

__device__ __forceinline__ unsigned short f2bf(float f) {
    unsigned int u = __float_as_uint(f);
    u += 0x7FFFu + ((u >> 16) & 1u);       // round-to-nearest-even
    return (unsigned short)(u >> 16);
}
__device__ __forceinline__ float sigm(float x) {
    return 1.0f / (1.0f + __expf(-x));
}
__device__ __forceinline__ float tanh_fast(float x) {
    x = fminf(15.0f, fmaxf(-15.0f, x));
    float e = __expf(2.0f * x);
    return (e - 1.0f) / (e + 1.0f);
}

// Split grid barrier: arrive ASAP after publishing h, wait as late as possible.
__device__ __forceinline__ void grid_arrive(unsigned* bar) {
    __syncthreads();                       // all threads' hb stores issued
    if (threadIdx.x == 0) {
        __threadfence();                   // release to device scope
        __hip_atomic_fetch_add(bar, 1u, __ATOMIC_RELEASE, __HIP_MEMORY_SCOPE_AGENT);
    }
}
__device__ __forceinline__ void grid_wait(unsigned* bar, unsigned target) {
    if (threadIdx.x == 0) {
        while (__hip_atomic_load(bar, __ATOMIC_ACQUIRE, __HIP_MEMORY_SCOPE_AGENT) < target) {
            __builtin_amdgcn_s_sleep(1);
        }
    }
    __syncthreads();
}

union Frag { uint4 q[2]; bf16x16 v; };

__global__ void __launch_bounds__(NTHR, 1)
lstm_init_kernel(unsigned* bar) {
    if (threadIdx.x < 64) bar[threadIdx.x] = 0u;
}

__global__ void __launch_bounds__(NTHR, 1)
lstm_persistent_kernel(const float* __restrict__ x,
                       const float* __restrict__ done,
                       const float* __restrict__ h0,
                       const float* __restrict__ c0,
                       const float* __restrict__ w_ih,
                       const float* __restrict__ w_hh,
                       const float* __restrict__ b_ih,
                       const float* __restrict__ b_hh,
                       float* __restrict__ out,
                       unsigned short* __restrict__ hb0,
                       unsigned short* __restrict__ hb1,
                       unsigned* __restrict__ bar) {
    extern __shared__ char smem[];
    unsigned short* bswu = (unsigned short*)smem;                      // weights bf16
    unsigned short* asu  = (unsigned short*)(smem + BSW_BYTES);        // A panel bf16
    float* gs   = (float*)(smem + BSW_BYTES + AS_BYTES);               // gates f32
    float* cs   = (float*)(smem + BSW_BYTES + AS_BYTES + GS_BYTES);    // cell state
    float* bsum = (float*)(smem + BSW_BYTES + AS_BYTES + GS_BYTES + CS_BYTES);
    float* dm   = bsum + 64;   // mask m_t
    float* mn   = dm + 64;     // mask m_{t+1}

    const int tid = threadIdx.x;
    const int wg  = blockIdx.x;            // 0..31 -> hidden cols [wg*16, wg*16+16)
    const int ln  = tid & 31;              // wave32 lane
    const int wv  = tid >> 5;              // wave 0..7
    const int rt  = wv & 3;                // row tile (batch)
    const int ct0 = (wv >> 2) * 2;         // col tiles ct0, ct0+1
    const int mrow  = ln & 15;
    const int khalf = ln >> 4;
    const int arow  = (rt << 4) + mrow;
    const int br0   = (ct0 << 4) + mrow;

    // ---- one-time init: weights -> LDS bf16, biases, cell state, h0 bcast ----
    for (int e = 0; e < 64; ++e) {
        int idx = (e << 10) + (tid << 2);          // 64*1024 elems, 4/thread/iter
        int row = idx >> 10;                       // local gate col 0..63
        int col = idx & 1023;                      // k 0..1023
        int grow = (row >> 4) * HID + (wg << 4) + (row & 15);
        const float* src = (col < 512) ? (w_ih + (size_t)grow * 512 + col)
                                       : (w_hh + (size_t)grow * 512 + (col - 512));
        float4 v = *(const float4*)src;
        ushort4 o; o.x = f2bf(v.x); o.y = f2bf(v.y); o.z = f2bf(v.z); o.w = f2bf(v.w);
        *(ushort4*)(bswu + row * WSTRIDE + col) = o;
    }
    if (tid < 64) {
        int grow = (tid >> 4) * HID + (wg << 4) + (tid & 15);
        bsum[tid] = b_ih[grow] + b_hh[grow];
    }
    for (int e = 0; e < 4; ++e) {
        int idx = (e << 8) + tid;                  // 1024 = 64 batch x 16 cols
        int b = idx >> 4, j = idx & 15;
        int col = (wg << 4) + j;
        cs[b * 16 + j] = c0[b * HID + col];
        float m0 = 1.0f - done[b];
        hb0[b * HID + col] = (m0 != 0.0f) ? f2bf(h0[b * HID + col]) : (unsigned short)0;
    }
    grid_arrive(bar);                              // phase 0: h for step 0 ready

    // ---- time loop ----
    for (int t = 0; t < TSTEPS; ++t) {
        const unsigned short* hbcur = (t & 1) ? hb1 : hb0;
        unsigned short*       hbnxt = (t & 1) ? hb0 : hb1;

        if (tid < 64) {
            dm[tid] = 1.0f - done[(size_t)t * BATCH + tid];
            mn[tid] = (t + 1 < TSTEPS) ? (1.0f - done[(size_t)(t + 1) * BATCH + tid]) : 0.0f;
        }

        // ---- stage x_t half of A panel (cols 0..511); no h dependence ----
        for (int e = 0; e < 32; ++e) {
            int idx = (e << 10) + (tid << 2);      // 64x512 elems
            int row = idx >> 9, col = idx & 511;
            float4 v = *(const float4*)(x + ((size_t)t * BATCH + row) * 512 + col);
            ushort4 o; o.x=f2bf(v.x); o.y=f2bf(v.y); o.z=f2bf(v.z); o.w=f2bf(v.w);
            *(ushort4*)(asu + row * ASTRIDE + col) = o;
        }
        __syncthreads();

        f32x8 acc0 = {0.f,0.f,0.f,0.f,0.f,0.f,0.f,0.f};
        f32x8 acc1 = {0.f,0.f,0.f,0.f,0.f,0.f,0.f,0.f};

        // ---- x-half GEMM: K = 0..511 (overlaps the barrier-wait window) ----
        #pragma unroll
        for (int kk = 0; kk < 16; ++kk) {
            const int kg = (kk << 5);
            Frag a, bA, bB;
            a.q[0]  = *(const uint4*)(asu  + arow * ASTRIDE + kg + (khalf << 3));
            a.q[1]  = *(const uint4*)(asu  + arow * ASTRIDE + kg + 16 + (khalf << 3));
            bA.q[0] = *(const uint4*)(bswu + br0 * WSTRIDE + kg + (khalf << 3));
            bA.q[1] = *(const uint4*)(bswu + br0 * WSTRIDE + kg + 16 + (khalf << 3));
            bB.q[0] = *(const uint4*)(bswu + (br0 + 16) * WSTRIDE + kg + (khalf << 3));
            bB.q[1] = *(const uint4*)(bswu + (br0 + 16) * WSTRIDE + kg + 16 + (khalf << 3));
            acc0 = __builtin_amdgcn_wmma_f32_16x16x32_bf16(
                false, a.v, false, bA.v, (short)0, acc0, false, false);
            acc1 = __builtin_amdgcn_wmma_f32_16x16x32_bf16(
                false, a.v, false, bB.v, (short)0, acc1, false, false);
        }

        // ---- wait until every WG published h_{t-1} -> hbcur valid ----
        grid_wait(bar, (unsigned)(NWG * (t + 1)));

        // ---- stage masked h half (cols 512..1023); already bf16+pre-masked ----
        for (int e = 0; e < 32; ++e) {
            int idx = (e << 10) + (tid << 2);      // 64x512 elems
            int row = idx >> 9, col = idx & 511;
            ushort4 v = *(const ushort4*)(hbcur + row * HID + col);
            *(ushort4*)(asu + row * ASTRIDE + 512 + col) = v;
        }
        __syncthreads();

        // ---- h-half GEMM: K = 512..1023 (the true recurrent critical path) ----
        #pragma unroll
        for (int kk = 0; kk < 16; ++kk) {
            const int kg = 512 + (kk << 5);
            Frag a, bA, bB;
            a.q[0]  = *(const uint4*)(asu  + arow * ASTRIDE + kg + (khalf << 3));
            a.q[1]  = *(const uint4*)(asu  + arow * ASTRIDE + kg + 16 + (khalf << 3));
            bA.q[0] = *(const uint4*)(bswu + br0 * WSTRIDE + kg + (khalf << 3));
            bA.q[1] = *(const uint4*)(bswu + br0 * WSTRIDE + kg + 16 + (khalf << 3));
            bB.q[0] = *(const uint4*)(bswu + (br0 + 16) * WSTRIDE + kg + (khalf << 3));
            bB.q[1] = *(const uint4*)(bswu + (br0 + 16) * WSTRIDE + kg + 16 + (khalf << 3));
            acc0 = __builtin_amdgcn_wmma_f32_16x16x32_bf16(
                false, a.v, false, bA.v, (short)0, acc0, false, false);
            acc1 = __builtin_amdgcn_wmma_f32_16x16x32_bf16(
                false, a.v, false, bB.v, (short)0, acc1, false, false);
        }

        // ---- spill accumulators (C layout: VGPR v -> M = v + khalf*8, lane -> N) ----
        {
            const int gcol  = (ct0 << 4) + (ln & 15);
            const int grow0 = (rt << 4) + (khalf << 3);
            #pragma unroll
            for (int v = 0; v < 8; ++v) {
                gs[(grow0 + v) * GSTRIDE + gcol]      = acc0[v];
                gs[(grow0 + v) * GSTRIDE + gcol + 16] = acc1[v];
            }
        }
        __syncthreads();

        // ---- gate math (fp32) + h broadcast; arrive BEFORE output stores ----
        float hv[4], cv[4];
        #pragma unroll
        for (int e = 0; e < 4; ++e) {
            int idx = (e << 8) + tid;              // 64 batch x 16 cols
            int b = idx >> 4, j = idx & 15;
            float ig = sigm(gs[b * GSTRIDE + j]           + bsum[j]);
            float fg = sigm(gs[b * GSTRIDE + 16 + j]      + bsum[16 + j]);
            float gg = tanh_fast(gs[b * GSTRIDE + 32 + j] + bsum[32 + j]);
            float og = sigm(gs[b * GSTRIDE + 48 + j]      + bsum[48 + j]);
            float cold = cs[b * 16 + j] * dm[b];
            float cnew = fg * cold + ig * gg;
            cs[b * 16 + j] = cnew;
            float h = og * tanh_fast(cnew);
            hv[e] = h; cv[e] = cnew;
            hbnxt[b * HID + (wg << 4) + j] =
                (mn[b] != 0.0f) ? f2bf(h) : (unsigned short)0;
        }
        grid_arrive(bar);                          // unblock other WGPs ASAP

        #pragma unroll
        for (int e = 0; e < 4; ++e) {
            int idx = (e << 8) + tid;
            int b = idx >> 4, j = idx & 15;
            int col = (wg << 4) + j;
            out[((size_t)t * BATCH + b) * HID + col] = hv[e];
            if (t == TSTEPS - 1) {
                size_t base = (size_t)TSTEPS * BATCH * HID;
                out[base + b * HID + col]               = hv[e];   // hT
                out[base + BATCH * HID + b * HID + col] = cv[e];   // cT
            }
        }
    }
}

extern "C" void kernel_launch(void* const* d_in, const int* in_sizes, int n_in,
                              void* d_out, int out_size, void* d_ws, size_t ws_size,
                              hipStream_t stream) {
    const float* x    = (const float*)d_in[0];
    const float* done = (const float*)d_in[1];
    const float* h0   = (const float*)d_in[2];
    const float* c0   = (const float*)d_in[3];
    const float* w_ih = (const float*)d_in[4];
    const float* w_hh = (const float*)d_in[5];
    const float* b_ih = (const float*)d_in[6];
    const float* b_hh = (const float*)d_in[7];
    float* out = (float*)d_out;

    unsigned*       bar = (unsigned*)d_ws;
    unsigned short* hb0 = (unsigned short*)((char*)d_ws + 256);
    unsigned short* hb1 = (unsigned short*)((char*)d_ws + 256 + BATCH * HID * 2);

    (void)in_sizes; (void)n_in; (void)out_size; (void)ws_size;

    hipFuncSetAttribute((const void*)lstm_persistent_kernel,
                        hipFuncAttributeMaxDynamicSharedMemorySize,
                        (int)SMEM_BYTES);

    hipLaunchKernelGGL(lstm_init_kernel, dim3(1), dim3(NTHR), 0, stream, bar);
    hipLaunchKernelGGL(lstm_persistent_kernel, dim3(NWG), dim3(NTHR), SMEM_BYTES, stream,
                       x, done, h0, c0, w_ih, w_hh, b_ih, b_hh, out, hb0, hb1, bar);
}
